// GPTAttention_50130858279255
// MI455X (gfx1250) — matrix operand order repaired and checked by reading, not compiled, not run on hardware
//
#include <hip/hip_runtime.h>
#include <hip/hip_bf16.h>
#include <math.h>

// ---------------------------------------------------------------------------
// CDNA5 WMMA types + helpers (wave32, v_wmma_f32_16x16x32_bf16)
// ---------------------------------------------------------------------------
typedef __attribute__((ext_vector_type(16))) __bf16 v16bf;
typedef __attribute__((ext_vector_type(8)))  float  v8f;
typedef unsigned int u32;
typedef unsigned long long u64;
typedef u32 __attribute__((ext_vector_type(4))) u32x4;
typedef int __attribute__((ext_vector_type(8))) i32x8;
typedef int __attribute__((ext_vector_type(4))) i32x4;

#if defined(__gfx1250__) && __has_builtin(__builtin_amdgcn_tensor_load_to_lds) && \
    __has_builtin(__builtin_amdgcn_s_wait_tensorcnt)
#define HAVE_TDM 1
#else
#define HAVE_TDM 0
#endif

__device__ __forceinline__ v8f zero8() {
    v8f z;
#pragma unroll
    for (int i = 0; i < 8; ++i) z[i] = 0.0f;
    return z;
}

__device__ __forceinline__ v8f wmma_bf16(v16bf a, v16bf b, v8f c) {
    // (neg_a, A, neg_b, B, c_mod, C, reuse_a, reuse_b)
    return __builtin_amdgcn_wmma_f32_16x16x32_bf16(false, a, false, b,
                                                   (short)0, c, false, false);
}

// A-fragment: 16(M) x 32(K) bf16, tile row-major in LDS, leading dim ld.
// Lane l: row M = l%16, half = l/16; element e -> K = 16*half + e.
__device__ __forceinline__ v16bf load_A_frag(const __bf16* tile, int ld) {
    const int lane = threadIdx.x & 31;
    const int row  = lane & 15;
    const int half = lane >> 4;
    const __bf16* p = tile + row * ld;
    v16bf a;
#pragma unroll
    for (int e = 0; e < 16; ++e) {
        int k = (half << 4) + e;
        a[e] = p[k];
    }
    return a;
}

// B-fragment: 32(K) x 16(N) bf16. tileT stored [N][K] (contraction contiguous).
// Lane l: column N = l%16, half = l/16; element e -> K = 16*half + e.
__device__ __forceinline__ v16bf load_B_frag(const __bf16* tileT, int ld) {
    const int lane = threadIdx.x & 31;
    const int n    = lane & 15;
    const int half = lane >> 4;
    const __bf16* p = tileT + n * ld + (half << 4);
    v16bf b;
#pragma unroll
    for (int e = 0; e < 16; ++e) b[e] = p[e];
    return b;
}

#if HAVE_TDM
// TDM: async DMA of a 2-D bf16 tile (tile_dim0=64 contiguous elems x `rows`
// rows, row stride `stride` elems) from global memory into LDS at lds_off.
// D# layout per CDNA5 ISA (group0 + group1;
// groups 2/3 zero => <=2D tensor). Tracked by TENSORcnt.
// This toolchain exposes the 6-arg builtin: (g0, g1, g2, g3, g_extra, cpol).
__device__ __forceinline__ void tdm_load_2d_bf16(const __bf16* gsrc, u32 lds_off,
                                                 u32 rows, u32 stride) {
    const u64 ga = (u64)(uintptr_t)gsrc;
    u32x4 g0;
    g0[0] = 1u;                                           // count=1, user D#
    g0[1] = lds_off;                                      // lds_addr (bytes)
    g0[2] = (u32)ga;                                      // global_addr[31:0]
    g0[3] = (u32)((ga >> 32) & 0x01FFFFFFu) | (2u << 30); // addr[56:32] | type=2
    i32x8 g1;
    g1[0] = (int)(1u << 16);              // workgroup_mask=0, data_size=1 (2B)
    g1[1] = (int)(64u << 16);             // tensor_dim0 = 64 (lo16 @bits63:48)
    g1[2] = 0;                            // dim0 hi16 = 0, tensor_dim1 lo16 = 0
    g1[3] = (int)(16u | (64u << 16));     // tensor_dim1 = 1<<20 (hi16), tile_dim0=64
    g1[4] = (int)rows;                    // tile_dim1 = rows, tile_dim2 = 0
    g1[5] = (int)stride;                  // tensor_dim0_stride lo32
    g1[6] = 0;                            // stride hi16, dim1_stride lo16
    g1[7] = 0;
    i32x4 z4 = {0, 0, 0, 0};
    i32x8 z8 = {0, 0, 0, 0, 0, 0, 0, 0};
    __builtin_amdgcn_tensor_load_to_lds(g0, g1, z4, z4, z8, 0);
}
#endif

// ---------------------------------------------------------------------------
// Generic GEMM + bias:  Cout[M,N] = A[M,K] @ W[K,N] + bias[N]
// Block: 128 threads (4 waves), tile 64(M) x 64(N), K-step 32.
// Wave w computes rows [16w,16w+16) x all 64 columns (4 WMMA accumulators).
// ---------------------------------------------------------------------------
template <typename TA, bool OUT_BF16>
__global__ __launch_bounds__(128)
void gemm_bias_kernel(const TA* __restrict__ A, const float* __restrict__ W,
                      const float* __restrict__ bias, void* __restrict__ Cout,
                      int M, int K, int N) {
    __shared__ __bf16 lA[64 * 32];   // [m_local][k]
    __shared__ __bf16 lBT[64 * 32];  // [n_local][k]  (transposed for B frags)

    const int tid  = threadIdx.x;
    const int lane = tid & 31;
    const int wv   = tid >> 5;
    const int half = lane >> 4;
    const int col  = lane & 15;
    const int n0   = blockIdx.x * 64;
    const int m0   = blockIdx.y * 64;

    // Per-thread fixed staging coordinates (strength-reduced pointers).
    const int ac  = (tid & 7) * 4;   // A: 4 consecutive k
    const int ar0 = tid >> 3;        // A: rows ar0, ar0+16, +32, +48
    const int wj  = (tid & 15) * 4;  // W: 4 consecutive n
    const int wc0 = tid >> 4;        // W: k-rows wc0, wc0+8, +16, +24

    v8f acc[4];
#pragma unroll
    for (int t = 0; t < 4; ++t) acc[t] = zero8();

    for (int kk = 0; kk < K; kk += 32) {
        // Stage A tile (64x32) as bf16 via 16-byte (f32) / 8-byte (bf16) loads.
        {
            const TA* pa = A + (size_t)(m0 + ar0) * K + kk + ac;
#pragma unroll
            for (int i = 0; i < 4; ++i) {
                union { uint2 u; __bf16 hh[4]; } cv;
                if constexpr (sizeof(TA) == 4) {
                    float4 f = *(const float4*)pa;
                    cv.hh[0] = (__bf16)f.x; cv.hh[1] = (__bf16)f.y;
                    cv.hh[2] = (__bf16)f.z; cv.hh[3] = (__bf16)f.w;
                } else {
                    cv.u = *(const uint2*)pa;
                }
                *(uint2*)&lA[(ar0 + 16 * i) * 32 + ac] = cv.u;
                pa += (size_t)16 * K;
            }
        }
        // Stage W tile (32x64) transposed -> [n][k] via float4 loads.
        {
            const float* pw = W + (size_t)(kk + wc0) * N + n0 + wj;
#pragma unroll
            for (int i = 0; i < 4; ++i) {
                float4 f = *(const float4*)pw;
                int c = wc0 + 8 * i;
                lBT[(wj + 0) * 32 + c] = (__bf16)f.x;
                lBT[(wj + 1) * 32 + c] = (__bf16)f.y;
                lBT[(wj + 2) * 32 + c] = (__bf16)f.z;
                lBT[(wj + 3) * 32 + c] = (__bf16)f.w;
                pw += (size_t)8 * N;
            }
        }
        if (kk + 32 < K) {  // prefetch next k-tile (global_prefetch_b8)
            __builtin_prefetch(A + (size_t)(m0 + ar0) * K + kk + 32 + ac, 0, 0);
            __builtin_prefetch(W + (size_t)(kk + 32 + wc0) * N + n0 + wj, 0, 0);
        }
        __syncthreads();

        v16bf a = load_A_frag(lA + wv * 16 * 32, 32);
#pragma unroll
        for (int t = 0; t < 4; ++t) {
            v16bf b = load_B_frag(lBT + t * 16 * 32, 32);
            acc[t] = wmma_bf16(a, b, acc[t]);
        }
        __syncthreads();
    }

    // Epilogue: bias + store. C layout: row = v + 8*half, col = lane&15.
    const int rowl = wv * 16 + half * 8;
#pragma unroll
    for (int t = 0; t < 4; ++t) {
        int cg = n0 + t * 16 + col;
        float bv = bias[cg];
#pragma unroll
        for (int v = 0; v < 8; ++v) {
            int rg = m0 + rowl + v;
            float val = acc[t][v] + bv;
            if constexpr (OUT_BF16)
                ((__bf16*)Cout)[(size_t)rg * N + cg] = (__bf16)val;
            else
                ((float*)Cout)[(size_t)rg * N + cg] = val;
        }
    }
}

// ---------------------------------------------------------------------------
// Flash attention (causal).  qkv: bf16 [B*T, 3C], Q|K|V at col 0|C|2C,
// head h at cols h*64..h*64+63 of each third.
// Block: 128 threads (4 waves).  grid = (T/64, B*H).
// Wave w owns query rows q0+16w..+15; loops over 32-key blocks with online
// softmax.  K block staged by the Tensor Data Mover (one 64x8 slice per
// wave) when available; V staged manually (needs transpose).
// ---------------------------------------------------------------------------
__global__ __launch_bounds__(128)
void attn_kernel(const __bf16* __restrict__ qkv, __bf16* __restrict__ out) {
    constexpr int T = 2048, C = 768, C3 = 2304, H = 12;

    __shared__ __bf16 lQ[64 * 64];    // [q_local][d], pre-scaled by 1/sqrt(D)
    __shared__ __bf16 lK[32 * 64];    // [k_local][d]   ([N][contraction] for S)
    __shared__ __bf16 lVT[64 * 32];   // [d][k_local]   ([N][contraction] for PV)
    __shared__ __bf16 lP[4][16 * 32]; // per-wave P relayout scratch

    const int tid  = threadIdx.x;
    const int lane = tid & 31;
    const int wv   = tid >> 5;
    const int half = lane >> 4;
    const int col  = lane & 15;

    const int q0 = blockIdx.x * 64;
    const int b  = blockIdx.y / H;
    const int h  = blockIdx.y % H;
    const size_t base = (size_t)b * T * C3;
    const int qoff = h * 64, koff = C + h * 64, voff = 2 * C + h * 64;

    // Per-thread fixed staging coords: 4 consecutive d, rows r0 + 8*i.
    const int dv = (tid & 15) * 4;
    const int r0 = tid >> 4;  // 0..7

    // Stage Q (scaled by 1/sqrt(64)) once, 8-byte vector loads.
    {
        const __bf16* pq = qkv + base + (size_t)(q0 + r0) * C3 + qoff + dv;
#pragma unroll
        for (int i = 0; i < 8; ++i) {
            union { uint2 u; __bf16 hh[4]; } cv;
            cv.u = *(const uint2*)pq;
#pragma unroll
            for (int j = 0; j < 4; ++j)
                cv.hh[j] = (__bf16)((float)cv.hh[j] * 0.125f);
            *(uint2*)&lQ[(r0 + 8 * i) * 64 + dv] = cv.u;
            pq += (size_t)8 * C3;
        }
    }
    __syncthreads();

    const int qw = q0 + wv * 16;  // first query row of this wave
    v16bf aq0 = load_A_frag(lQ + wv * 16 * 64, 64);       // d 0..31
    v16bf aq1 = load_A_frag(lQ + wv * 16 * 64 + 32, 64);  // d 32..63

    v8f o[4];
#pragma unroll
    for (int t = 0; t < 4; ++t) o[t] = zero8();
    float mrow[8], lrow[8];
#pragma unroll
    for (int v = 0; v < 8; ++v) { mrow[v] = -INFINITY; lrow[v] = 0.0f; }

#if HAVE_TDM
    const u32 lK_off = (u32)(uintptr_t)(&lK[0]);  // LDS byte offset (low 32 bits)
#endif

    const int nkb = q0 / 32 + 2;  // key blocks up to and including diagonal
    for (int kbi = 0; kbi < nkb; ++kbi) {
        const int kb = kbi * 32;

        // ---- Stage K block (32x64, row-major) ----
#if HAVE_TDM
        // Each wave DMAs its own 8-row slice via the Tensor Data Mover.
        tdm_load_2d_bf16(qkv + base + (size_t)(kb + wv * 8) * C3 + koff,
                         lK_off + (u32)(wv * 8 * 64 * 2), 8u, (u32)C3);
#else
        {
            const __bf16* pk = qkv + base + (size_t)(kb + r0) * C3 + koff + dv;
#pragma unroll
            for (int i = 0; i < 4; ++i) {
                *(uint2*)&lK[(r0 + 8 * i) * 64 + dv] = *(const uint2*)pk;
                pk += (size_t)8 * C3;
            }
        }
#endif
        // ---- Stage V block transposed -> [d][key] ----
        {
            const __bf16* pv = qkv + base + (size_t)(kb + r0) * C3 + voff + dv;
#pragma unroll
            for (int i = 0; i < 4; ++i) {
                union { uint2 u; __bf16 hh[4]; } cv;
                cv.u = *(const uint2*)pv;
                int r = r0 + 8 * i;
#pragma unroll
                for (int j = 0; j < 4; ++j) lVT[(dv + j) * 32 + r] = cv.hh[j];
                pv += (size_t)8 * C3;
            }
        }
        if (kbi + 1 < nkb) {  // prefetch next block (global_prefetch_b8)
            __builtin_prefetch(qkv + base + (size_t)(kb + 32 + r0) * C3 + koff + dv, 0, 0);
            __builtin_prefetch(qkv + base + (size_t)(kb + 32 + r0) * C3 + voff + dv, 0, 0);
        }
#if HAVE_TDM
        __builtin_amdgcn_s_wait_tensorcnt(0);  // this wave's K slice landed
#endif
        __syncthreads();  // all slices visible to all waves

        if (kb <= qw + 15) {  // wave-uniform: block not fully above diagonal
            // S = Q @ K^T : two 16x16 C-tiles (keys kb..+15, kb+16..+31)
            v8f s0 = zero8(), s1 = zero8();
            {
                v16bf b0 = load_B_frag(lK, 64);
                v16bf b1 = load_B_frag(lK + 32, 64);
                s0 = wmma_bf16(aq0, b0, s0);
                s0 = wmma_bf16(aq1, b1, s0);
                v16bf b2 = load_B_frag(lK + 16 * 64, 64);
                v16bf b3 = load_B_frag(lK + 16 * 64 + 32, 64);
                s1 = wmma_bf16(aq0, b2, s1);
                s1 = wmma_bf16(aq1, b3, s1);
            }
            // Causal mask (only needed near the diagonal).
            if (kb + 31 > qw) {
#pragma unroll
                for (int v = 0; v < 8; ++v) {
                    int rg = qw + half * 8 + v;
                    if (kb + col > rg)      s0[v] = -INFINITY;
                    if (kb + 16 + col > rg) s1[v] = -INFINITY;
                }
            }
            // Online softmax: row reductions across the 16-lane half-wave.
            float scl[8];
#pragma unroll
            for (int v = 0; v < 8; ++v) {
                float mx = fmaxf(s0[v], s1[v]);
#pragma unroll
                for (int off = 1; off < 16; off <<= 1)
                    mx = fmaxf(mx, __shfl_xor(mx, off, 16));
                float mnew = fmaxf(mrow[v], mx);
                scl[v]  = __expf(mrow[v] - mnew);
                mrow[v] = mnew;
                float p0 = __expf(s0[v] - mnew);
                float p1 = __expf(s1[v] - mnew);
                s0[v] = p0; s1[v] = p1;
                float sum = p0 + p1;
#pragma unroll
                for (int off = 1; off < 16; off <<= 1)
                    sum += __shfl_xor(sum, off, 16);
                lrow[v] = lrow[v] * scl[v] + sum;
            }
            // Relayout P (C-layout) -> A-fragment via per-wave LDS; rescale O.
            __bf16* Pw = lP[wv];
#pragma unroll
            for (int v = 0; v < 8; ++v) {
                int r = half * 8 + v;
                Pw[r * 32 + col]      = (__bf16)s0[v];
                Pw[r * 32 + 16 + col] = (__bf16)s1[v];
                o[0][v] *= scl[v]; o[1][v] *= scl[v];
                o[2][v] *= scl[v]; o[3][v] *= scl[v];
            }
            asm volatile("s_wait_dscnt 0" ::: "memory");  // wave-local LDS RAW

            // O += P @ V  (contraction over 32 keys, 4 d-tiles of 16)
            v16bf ap = load_A_frag(Pw, 32);
#pragma unroll
            for (int t = 0; t < 4; ++t) {
                v16bf bvf = load_B_frag(lVT + t * 16 * 32, 32);
                o[t] = wmma_bf16(ap, bvf, o[t]);
            }
        }
        __syncthreads();  // before next block overwrites lK/lVT
    }

    // Normalize and store: out[b*T + q][h*64 + d], bf16.
#pragma unroll
    for (int t = 0; t < 4; ++t) {
#pragma unroll
        for (int v = 0; v < 8; ++v) {
            int rg = qw + half * 8 + v;
            int d  = t * 16 + col;
            out[(size_t)(b * T + rg) * C + h * 64 + d] =
                (__bf16)(o[t][v] / lrow[v]);
        }
    }
}

// ---------------------------------------------------------------------------
// Launch: GEMM(qkv) -> flash attention -> GEMM(proj)
// ws layout: qkv bf16 [4096,2304] (18.9 MB) | att bf16 [4096,768] (6.3 MB)
// ---------------------------------------------------------------------------
extern "C" void kernel_launch(void* const* d_in, const int* in_sizes, int n_in,
                              void* d_out, int out_size, void* d_ws, size_t ws_size,
                              hipStream_t stream) {
    const float* x  = (const float*)d_in[0];  // [2,2048,768]
    const float* w1 = (const float*)d_in[1];  // [768,2304]
    const float* b1 = (const float*)d_in[2];  // [2304]
    const float* w2 = (const float*)d_in[3];  // [768,768]
    const float* b2 = (const float*)d_in[4];  // [768]

    const int B = 2, T = 2048, C = 768, M = B * T;

    __bf16* qkv = (__bf16*)d_ws;
    __bf16* att = (__bf16*)((char*)d_ws + (size_t)M * 3 * C * sizeof(__bf16));

    dim3 blk(128);
    gemm_bias_kernel<float, true>
        <<<dim3((3 * C) / 64, M / 64), blk, 0, stream>>>(x, w1, b1, qkv, M, C, 3 * C);
    attn_kernel<<<dim3(T / 64, B * 12), blk, 0, stream>>>(qkv, att);
    gemm_bias_kernel<__bf16, false>
        <<<dim3(C / 64, M / 64), blk, 0, stream>>>(att, w2, b2, d_out, M, C, C);
}
